// LuongAttnDecoderRNN_74199855006055
// MI455X (gfx1250) — matrix operand order repaired and checked by reading, not compiled, not run on hardware
//
#include <hip/hip_runtime.h>
#include <hip/hip_bf16.h>

// ---------------- problem constants ----------------
constexpr int Bc  = 64;
constexpr int Sc  = 512;
constexpr int Hc  = 512;
constexpr int Ec  = 512;
constexpr int H2  = 1024;   // 2*H
constexpr int H3  = 1536;   // 3*H
constexpr int Oc  = 128;
constexpr int OPN = 4;

// ---------------- vector types ----------------
typedef __attribute__((ext_vector_type(16))) __bf16 bf16x16;
typedef __attribute__((ext_vector_type(8)))  __bf16 bf16x8;
typedef __attribute__((ext_vector_type(4)))  __bf16 bf16x4;
typedef __attribute__((ext_vector_type(8)))  float  f32x8;

// ---------------- block reductions ----------------
__device__ inline float redSum256(float v) {
    __shared__ float sh[256];
    int t = threadIdx.x;
    sh[t] = v; __syncthreads();
    #pragma unroll
    for (int o = 128; o > 0; o >>= 1) { if (t < o) sh[t] += sh[t + o]; __syncthreads(); }
    float r = sh[0]; __syncthreads(); return r;
}
__device__ inline float redMax256(float v) {
    __shared__ float sh[256];
    int t = threadIdx.x;
    sh[t] = v; __syncthreads();
    #pragma unroll
    for (int o = 128; o > 0; o >>= 1) { if (t < o) sh[t] = fmaxf(sh[t], sh[t + o]); __syncthreads(); }
    float r = sh[0]; __syncthreads(); return r;
}
__device__ inline float redSum128(float v) {
    __shared__ float sh[128];
    int t = threadIdx.x;
    sh[t] = v; __syncthreads();
    #pragma unroll
    for (int o = 64; o > 0; o >>= 1) { if (t < o) sh[t] += sh[t + o]; __syncthreads(); }
    float r = sh[0]; __syncthreads(); return r;
}
__device__ inline float redMax128(float v) {
    __shared__ float sh[128];
    int t = threadIdx.x;
    sh[t] = v; __syncthreads();
    #pragma unroll
    for (int o = 64; o > 0; o >>= 1) { if (t < o) sh[t] = fmaxf(sh[t], sh[t + o]); __syncthreads(); }
    float r = sh[0]; __syncthreads(); return r;
}

// ---------------- WMMA helpers (CDNA5 wave32, 16x16x32 bf16) ----------------
__device__ inline f32x8 wmma_bf16(bf16x16 a, bf16x16 b, f32x8 c) {
    return __builtin_amdgcn_wmma_f32_16x16x32_bf16(false, a, false, b, (short)0, c, false, false);
}

// A-fragment (16x32, row-major in memory). Per ISA layout:
// lanes 0-15 : M=lane,   elems 0..7 -> K=0..7,   elems 8..15 -> K=16..23
// lanes 16-31: M=lane-16,elems 0..7 -> K=8..15,  elems 8..15 -> K=24..31
// => lane loads 8 bf16 at (row=lane&15, col=(lane>>4)*8) and at col+16.
__device__ inline bf16x16 load_fragA(const __bf16* __restrict__ p /* + row*lda + (lane>>4)*8 */) {
    bf16x8 lo = *(const bf16x8*)p;
    bf16x8 hi = *(const bf16x8*)(p + 16);
    return __builtin_shufflevector(lo, hi, 0,1,2,3,4,5,6,7,8,9,10,11,12,13,14,15);
}
// B-fragment from Bt (N x K row-major, i.e. B transposed):
// lane covers N=lane&15, K = (lane>>4)*8 .. +7 and +16 .. +23 (same as A).
__device__ inline bf16x16 load_fragB(const __bf16* __restrict__ p /* + n*ldb + (lane>>4)*8 */) {
    bf16x8 lo = *(const bf16x8*)p;
    bf16x8 hi = *(const bf16x8*)(p + 16);
    return __builtin_shufflevector(lo, hi, 0,1,2,3,4,5,6,7,8,9,10,11,12,13,14,15);
}

// ---------------- GEMM with fused epilogues ----------------
// C[m,n] = sum_k A[m,k] * Bt[n,k]
// Tile: block 256 thr = 8 waves (2 x 4); wave tile = 64x32 (8 WMMAs / 6 frag loads per k-step)
// Block tile = 128 x 128.
// EPI 0: XW1T   : m = s*Bc+b ; out bf16 XW1t[b][n][s] = acc + aw[b,s]*P[b,n]
// EPI 1: H1     : batched z=b; out bf16 h1[b][m][n]   = relu(acc + bias[n])
// EPI 2: Y3T    : m = b*Sc+s ; out bf16 Y3t[b][n][s]  = acc            (n < H2)
// EPI 3: W2W    : batched z=b; out bf16 w2w[b][m][n]  = acc + bias[n]  (n < H2)
template <int EPI>
__global__ __launch_bounds__(256) void k_gemm(
    const __bf16* __restrict__ Abase, const __bf16* __restrict__ Btbase,
    int Kdim, int lda, int ldb, size_t aStride, size_t bStride,
    __bf16* __restrict__ outB, const float* __restrict__ bias,
    const float* __restrict__ aw, const float* __restrict__ Pm)
{
    const int z = blockIdx.z;
    const __bf16* A  = Abase  + (size_t)z * aStride;
    const __bf16* Bt = Btbase + (size_t)z * bStride;

    const int lane = threadIdx.x & 31;
    const int wave = threadIdx.x >> 5;
    const int wm = wave >> 2;          // 0..1
    const int wn = wave & 3;           // 0..3
    const int m0 = blockIdx.y * 128 + wm * 64;
    const int n0 = blockIdx.x * 128 + wn * 32;

    f32x8 acc[4][2];
    #pragma unroll
    for (int i = 0; i < 4; ++i)
        #pragma unroll
        for (int j = 0; j < 2; ++j)
            acc[i][j] = (f32x8){0,0,0,0,0,0,0,0};

    const int ar = lane & 15;
    const int ac = (lane >> 4) << 3;   // 0 or 8
    const int bn = lane & 15;
    const int bk = (lane >> 4) << 3;   // 0 or 8

    const __bf16* pa = A  + (size_t)(m0 + ar) * lda + ac;
    const __bf16* pb = Bt + (size_t)(n0 + bn) * ldb + bk;
    const size_t aRow16 = (size_t)16 * lda;
    const size_t bRow16 = (size_t)16 * ldb;

    for (int k = 0; k < Kdim; k += 32) {
        bf16x16 a0 = load_fragA(pa + k);
        bf16x16 a1 = load_fragA(pa + aRow16 + k);
        bf16x16 a2 = load_fragA(pa + 2 * aRow16 + k);
        bf16x16 a3 = load_fragA(pa + 3 * aRow16 + k);
        bf16x16 b0 = load_fragB(pb + k);
        bf16x16 b1 = load_fragB(pb + bRow16 + k);
        acc[0][0] = wmma_bf16(a0, b0, acc[0][0]);
        acc[0][1] = wmma_bf16(a0, b1, acc[0][1]);
        acc[1][0] = wmma_bf16(a1, b0, acc[1][0]);
        acc[1][1] = wmma_bf16(a1, b1, acc[1][1]);
        acc[2][0] = wmma_bf16(a2, b0, acc[2][0]);
        acc[2][1] = wmma_bf16(a2, b1, acc[2][1]);
        acc[3][0] = wmma_bf16(a3, b0, acc[3][0]);
        acc[3][1] = wmma_bf16(a3, b1, acc[3][1]);
    }

    // C layout per ISA: elem e of v8f -> M = e + (lane>>4)*8, N = lane&15
    const int mLane = (lane >> 4) * 8;
    const int nLane = lane & 15;
    #pragma unroll
    for (int i = 0; i < 4; ++i) {
        #pragma unroll
        for (int jj = 0; jj < 2; ++jj) {
            #pragma unroll
            for (int e = 0; e < 8; ++e) {
                const int m = m0 + i * 16 + mLane + e;
                const int n = n0 + jj * 16 + nLane;
                float v = acc[i][jj][e];
                if constexpr (EPI == 0) {
                    const int b = m & (Bc - 1), s = m >> 6;          // m = s*Bc + b
                    v += aw[b * Sc + s] * Pm[b * Hc + n];
                    outB[((size_t)b * Hc + n) * Sc + s] = (__bf16)v;
                } else if constexpr (EPI == 1) {
                    v += bias[n]; v = v > 0.f ? v : 0.f;
                    outB[((size_t)z * Sc + m) * Hc + n] = (__bf16)v;
                } else if constexpr (EPI == 2) {
                    const int b = m >> 9, s = m & (Sc - 1);          // m = b*Sc + s
                    outB[((size_t)b * H2 + n) * Sc + s] = (__bf16)v;
                } else {
                    v += bias[n];
                    outB[((size_t)z * Sc + m) * H2 + n] = (__bf16)v;
                }
            }
        }
    }
}

// ---------------- small kernels ----------------
__global__ void k_gih(const int* __restrict__ step, const float* __restrict__ emb,
                      const float* __restrict__ lh,
                      const float* __restrict__ Wih, const float* __restrict__ bih,
                      const float* __restrict__ Whh, const float* __restrict__ bhh,
                      float* __restrict__ gi, float* __restrict__ gh)
{
    int idx = blockIdx.x * blockDim.x + threadIdx.x;
    if (idx >= Bc * H3) return;
    int b = idx / H3, j = idx % H3;
    const float* x  = emb + (size_t)step[b] * Ec;
    const float* wi = Wih + (size_t)j * Ec;
    float s1 = bih[j];
    for (int e = 0; e < Ec; ++e) s1 += x[e] * wi[e];
    gi[idx] = s1;
    const float* h  = lh + (size_t)b * Hc;
    const float* wh = Whh + (size_t)j * Hc;
    float s2 = bhh[j];
    for (int e = 0; e < Hc; ++e) s2 += h[e] * wh[e];
    gh[idx] = s2;
}

__global__ void k_hnew(const float* __restrict__ gi, const float* __restrict__ gh,
                       const float* __restrict__ lh, float* __restrict__ hout)
{
    int idx = blockIdx.x * blockDim.x + threadIdx.x;
    if (idx >= Bc * Hc) return;
    int b = idx / Hc, i = idx % Hc;
    const float* gib = gi + (size_t)b * H3;
    const float* ghb = gh + (size_t)b * H3;
    float r = 1.f / (1.f + expf(-(gib[i] + ghb[i])));
    float z = 1.f / (1.f + expf(-(gib[Hc + i] + ghb[Hc + i])));
    float n = tanhf(gib[2 * Hc + i] + r * ghb[2 * Hc + i]);
    float hp = lh[(size_t)b * Hc + i];
    hout[idx] = (1.f - z) * n + z * hp;
}

// u[b,k] = sum_h hn[b,h]*attn_W[h,k] ; P[b,k] = sum_h hn[b,h]*gc1_W[h,k]
__global__ void k_uP(const float* __restrict__ hn, const float* __restrict__ attnW,
                     const float* __restrict__ gc1W, float* __restrict__ u, float* __restrict__ P)
{
    int idx = blockIdx.x * blockDim.x + threadIdx.x;
    if (idx >= Bc * Hc) return;
    int b = idx / Hc, k = idx % Hc;
    const float* hb = hn + (size_t)b * Hc;
    float s1 = 0.f, s2 = 0.f;
    for (int h = 0; h < Hc; ++h) {
        float hv = hb[h];
        s1 += hv * attnW[(size_t)h * Hc + k];
        s2 += hv * gc1W[(size_t)h * Hc + k];
    }
    u[idx] = s1; P[idx] = s2;
}

__global__ void k_c0(const float* __restrict__ hn, const float* __restrict__ ab,
                     float* __restrict__ c0)
{
    int b = blockIdx.x, t = threadIdx.x;
    float acc = 0.f;
    for (int h = t; h < Hc; h += 256) acc += hn[(size_t)b * Hc + h] * ab[h];
    float s = redSum256(acc);
    if (t == 0) c0[b] = s;
}

__global__ void k_scores(const float* __restrict__ u, const float* __restrict__ enc,
                         const float* __restrict__ c0, float* __restrict__ sc)
{
    int idx = blockIdx.x * blockDim.x + threadIdx.x;
    if (idx >= Bc * Sc) return;
    int b = idx / Sc, s = idx % Sc;
    const float* e  = enc + ((size_t)s * Bc + b) * Hc;
    const float* ub = u + (size_t)b * Hc;
    float acc = c0[b];
    for (int h = 0; h < Hc; ++h) acc += ub[h] * e[h];
    sc[idx] = acc;
}

__global__ void k_softmax(const float* __restrict__ sc, float* __restrict__ aw)
{
    int b = blockIdx.x, t = threadIdx.x;
    float v0 = sc[(size_t)b * Sc + t];
    float v1 = sc[(size_t)b * Sc + t + 256];
    float mx = redMax256(fmaxf(v0, v1));
    float e0 = expf(v0 - mx), e1 = expf(v1 - mx);
    float sum = redSum256(e0 + e1);
    aw[(size_t)b * Sc + t]       = e0 / sum;
    aw[(size_t)b * Sc + t + 256] = e1 / sum;
}

__global__ void k_context(const float* __restrict__ aw, const float* __restrict__ enc,
                          float* __restrict__ ctx)
{
    int idx = blockIdx.x * blockDim.x + threadIdx.x;
    if (idx >= Bc * Hc) return;
    int b = idx / Hc, h = idx % Hc;
    float acc = 0.f;
    for (int s = 0; s < Sc; ++s)
        acc += aw[(size_t)b * Sc + s] * enc[((size_t)s * Bc + b) * Hc + h];
    ctx[idx] = acc;
}

__global__ void k_concat(const float* __restrict__ hn, const float* __restrict__ ctx,
                         const float* __restrict__ cW, const float* __restrict__ cb,
                         float* __restrict__ out)
{
    int idx = blockIdx.x * blockDim.x + threadIdx.x;
    if (idx >= Bc * Hc) return;
    int b = idx / Hc, h = idx % Hc;
    const float* w = cW + (size_t)h * H2;
    const float* hb = hn + (size_t)b * Hc;
    const float* cx = ctx + (size_t)b * Hc;
    float acc = cb[h];
    for (int k = 0; k < Hc; ++k) acc += hb[k] * w[k] + cx[k] * w[Hc + k];
    out[idx] = acc > 0.f ? acc : 0.f;
}

__global__ void k_convE(const float* __restrict__ e, __bf16* __restrict__ o, int n4)
{
    int i = blockIdx.x * blockDim.x + threadIdx.x;
    if (i >= n4) return;
    float4 v = ((const float4*)e)[i];
    bf16x4 r; r[0] = (__bf16)v.x; r[1] = (__bf16)v.y; r[2] = (__bf16)v.z; r[3] = (__bf16)v.w;
    ((bf16x4*)o)[i] = r;
}

__global__ void k_adj(const float* __restrict__ ww, const int* __restrict__ ex,
                      __bf16* __restrict__ abf, int n4)
{
    int i = blockIdx.x * blockDim.x + threadIdx.x;
    if (i >= n4) return;
    float4 w = ((const float4*)ww)[i];
    int4   e = ((const int4*)ex)[i];
    bf16x4 r;
    r[0] = (e.x == 1) ? (__bf16)w.x : (__bf16)0.f;
    r[1] = (e.y == 1) ? (__bf16)w.y : (__bf16)0.f;
    r[2] = (e.z == 1) ? (__bf16)w.z : (__bf16)0.f;
    r[3] = (e.w == 1) ? (__bf16)w.w : (__bf16)0.f;
    ((bf16x4*)abf)[i] = r;
}

// W1t[n,k] = gc1_W[H+k, n]   (H x H)
__global__ void k_w1t(const float* __restrict__ gc1W, __bf16* __restrict__ w1t)
{
    int idx = blockIdx.x * blockDim.x + threadIdx.x;
    if (idx >= Hc * Hc) return;
    int n = idx / Hc, k = idx % Hc;
    w1t[idx] = (__bf16)gc1W[(size_t)(Hc + k) * Hc + n];
}
// W2t[n,k] = gc2_W[k, n]     (2H x H)
__global__ void k_w2t(const float* __restrict__ gc2W, __bf16* __restrict__ w2t)
{
    int idx = blockIdx.x * blockDim.x + threadIdx.x;
    if (idx >= H2 * Hc) return;
    int n = idx / Hc, k = idx % Hc;
    w2t[idx] = (__bf16)gc2W[(size_t)k * H2 + n];
}

// LayerNorm(w2w_pre)*g+b + w_all, in place on bf16 (B*S rows of 1024)
__global__ __launch_bounds__(256) void k_ln_w2w(__bf16* __restrict__ w2w,
    const float* __restrict__ g, const float* __restrict__ bb,
    const float* __restrict__ aw, const float* __restrict__ hn,
    const float* __restrict__ enc)
{
    int row = blockIdx.x;                 // b*Sc + s
    int b = row >> 9, s = row & (Sc - 1);
    __bf16* x = w2w + (size_t)row * H2;
    int t = threadIdx.x;
    float loc[4]; float sum = 0.f, sq = 0.f;
    #pragma unroll
    for (int i = 0; i < 4; ++i) {
        float v = (float)x[t + 256 * i];
        loc[i] = v; sum += v; sq += v * v;
    }
    float mean = redSum256(sum) * (1.f / H2);
    float var  = redSum256(sq) * (1.f / H2) - mean * mean;
    float rs = rsqrtf(var + 1e-6f);
    float awv = aw[(size_t)b * Sc + s];
    #pragma unroll
    for (int i = 0; i < 4; ++i) {
        int d = t + 256 * i;
        float res = (d < Hc) ? awv * hn[(size_t)b * Hc + d]
                             : enc[((size_t)s * Bc + b) * Hc + (d - Hc)];
        x[d] = (__bf16)((loc[i] - mean) * rs * g[d] + bb[d] + res);
    }
}

// wo_n[b,o,s] = masked word_op normalized over s
__global__ void k_won(const float* __restrict__ word_op, const int* __restrict__ seq_mask,
                      float* __restrict__ won)
{
    int row = blockIdx.x;                 // b*OPN + o
    int b = row >> 2, o = row & 3;
    int t = threadIdx.x;
    float* out = won + (size_t)row * Sc;
    float acc = 0.f;
    for (int s = t; s < Sc; s += 256) {
        float v = (seq_mask[(size_t)b * Sc + s] != 0)
                    ? 0.f : word_op[((size_t)b * Sc + s) * OPN + o];
        out[s] = v; acc += v;
    }
    float sum = redSum256(acc) + 1e-30f;
    for (int s = t; s < Sc; s += 256) out[s] /= sum;
}

// op1[b,o,d] = sum_s wo_n[b,o,s] * w2w[b,s,d]   (d < 2H)
__global__ void k_optrans1(const float* __restrict__ won, const __bf16* __restrict__ w2w,
                           float* __restrict__ op1)
{
    int idx = blockIdx.x * blockDim.x + threadIdx.x;
    if (idx >= Bc * OPN * H2) return;
    int d = idx % H2, o = (idx / H2) % OPN, b = idx / (H2 * OPN);
    const float* w = won + ((size_t)b * OPN + o) * Sc;
    const __bf16* m = w2w + (size_t)b * Sc * H2 + d;
    float acc = 0.f;
    for (int s = 0; s < Sc; ++s) acc += w[s] * (float)m[(size_t)s * H2];
    op1[idx] = acc;
}

// op2[b,o,h] = relu( op1[b,o,:] . wo_W[h,:] + wo_b[h] )
__global__ void k_optrans2(const float* __restrict__ op1, const float* __restrict__ woW,
                           const float* __restrict__ wob, float* __restrict__ op2)
{
    int idx = blockIdx.x * blockDim.x + threadIdx.x;
    if (idx >= Bc * OPN * Hc) return;
    int h = idx % Hc; int row = idx / Hc;
    const float* x = op1 + (size_t)row * H2;
    const float* w = woW + (size_t)h * H2;
    float acc = wob[h];
    for (int d = 0; d < H2; ++d) acc += x[d] * w[d];
    op2[idx] = acc > 0.f ? acc : 0.f;
}

// in-place LN over H=512, optional residual ops[o,:]
__global__ void k_ln_small(float* __restrict__ x, const float* __restrict__ g,
                           const float* __restrict__ bb, const float* __restrict__ resid)
{
    int row = blockIdx.x; int o = row & 3; int t = threadIdx.x;
    float* xr = x + (size_t)row * Hc;
    float v0 = xr[t], v1 = xr[t + 256];
    float mean = redSum256(v0 + v1) * (1.f / Hc);
    float d0 = v0 - mean, d1 = v1 - mean;
    float var = redSum256(d0 * d0 + d1 * d1) * (1.f / Hc);
    float rs = rsqrtf(var + 1e-6f);
    float r0 = resid ? resid[(size_t)o * Hc + t]       : 0.f;
    float r1 = resid ? resid[(size_t)o * Hc + t + 256] : 0.f;
    xr[t]       = d0 * rs * g[t]       + bb[t]       + r0;
    xr[t + 256] = d1 * rs * g[t + 256] + bb[t + 256] + r1;
}

// op3[b,o,h] = relu( op2 . ot_W[h,:H] + ops[o,:] . ot_W[h,H:] + ot_b[h] )
__global__ void k_oph(const float* __restrict__ op2, const float* __restrict__ ops,
                      const float* __restrict__ otW, const float* __restrict__ otb,
                      float* __restrict__ op3)
{
    int idx = blockIdx.x * blockDim.x + threadIdx.x;
    if (idx >= Bc * OPN * Hc) return;
    int h = idx % Hc; int row = idx / Hc; int o = row & 3;
    const float* x = op2 + (size_t)row * Hc;
    const float* p = ops + (size_t)o * Hc;
    const float* w = otW + (size_t)h * H2;
    float acc = otb[h];
    for (int d = 0; d < Hc; ++d) acc += x[d] * w[d] + p[d] * w[Hc + d];
    op3[idx] = acc > 0.f ? acc : 0.f;
}

__global__ void k_opout(const float* __restrict__ concat, const float* __restrict__ op3,
                        const float* __restrict__ ops_bias, float* __restrict__ logits)
{
    int row = blockIdx.x; int b = row >> 2, o = row & 3; int t = threadIdx.x;
    const float* c = concat + (size_t)b * Hc;
    const float* v = op3 + (size_t)row * Hc;
    float acc = c[t] * v[t] + c[t + 256] * v[t + 256];
    float s = redSum256(acc);
    if (t == 0) logits[(size_t)b * Oc + o] = s + ops_bias[o];
}

__global__ void k_nop(const float* __restrict__ concat, const float* __restrict__ nopW,
                      const float* __restrict__ nopb, float* __restrict__ logits)
{
    int idx = blockIdx.x * blockDim.x + threadIdx.x;
    if (idx >= Bc * (Oc - OPN)) return;
    int b = idx / (Oc - OPN), j = idx % (Oc - OPN);
    const float* c = concat + (size_t)b * Hc;
    const float* w = nopW + (size_t)j * Hc;
    float acc = nopb[j];
    for (int h = 0; h < Hc; ++h) acc += c[h] * w[h];
    logits[(size_t)b * Oc + OPN + j] = acc;
}

__global__ void k_logsm(const float* __restrict__ logits, float* __restrict__ out)
{
    int b = blockIdx.x, t = threadIdx.x;          // 128 threads
    float v = logits[(size_t)b * Oc + t];
    float mx = redMax128(v);
    float sum = redSum128(expf(v - mx));
    out[(size_t)b * Oc + t] = v - mx - logf(sum);
}

// ---------------- launcher ----------------
extern "C" void kernel_launch(void* const* d_in, const int* in_sizes, int n_in,
                              void* d_out, int out_size, void* d_ws, size_t ws_size,
                              hipStream_t stream)
{
    (void)in_sizes; (void)n_in; (void)out_size; (void)ws_size;
    const int*   step  = (const int*)  d_in[0];
    const float* lh    = (const float*)d_in[1];
    const float* enc   = (const float*)d_in[2];
    const float* ww    = (const float*)d_in[3];
    const float* wop   = (const float*)d_in[4];
    const int*   wex   = (const int*)  d_in[5];
    const int*   smask = (const int*)  d_in[6];
    const float* emb   = (const float*)d_in[7];
    const float* Wih   = (const float*)d_in[8];
    const float* Whh   = (const float*)d_in[9];
    const float* bih   = (const float*)d_in[10];
    const float* bhh   = (const float*)d_in[11];
    const float* attnW = (const float*)d_in[12];
    const float* attnb = (const float*)d_in[13];
    const float* cW    = (const float*)d_in[14];
    const float* cb    = (const float*)d_in[15];
    const float* ops   = (const float*)d_in[16];
    const float* opsb  = (const float*)d_in[17];
    const float* nopW  = (const float*)d_in[18];
    const float* nopb  = (const float*)d_in[19];
    const float* gc1W  = (const float*)d_in[20];
    const float* gc1b  = (const float*)d_in[21];
    const float* gc2W  = (const float*)d_in[22];
    const float* gc2b  = (const float*)d_in[23];
    const float* ng    = (const float*)d_in[24];
    const float* nb    = (const float*)d_in[25];
    const float* n1g   = (const float*)d_in[26];
    const float* n1b   = (const float*)d_in[27];
    const float* n2g   = (const float*)d_in[28];
    const float* n2b   = (const float*)d_in[29];
    const float* woW   = (const float*)d_in[30];
    const float* wob   = (const float*)d_in[31];
    const float* otW   = (const float*)d_in[32];
    const float* otb   = (const float*)d_in[33];

    float* out = (float*)d_out;
    float* out_logits = out;                 // (B, O)
    float* out_h      = out + 8192;          // (1, B, H)
    float* out_aw     = out + 40960;         // (B, 1, S)
    float* out_concat = out + 73728;         // (B, H)

    // fp32 scratch pool
    float* wsf = (float*)d_ws;
    float* GI  = wsf;                        // B*3H
    float* GH  = GI  + 98304;                // B*3H
    float* U   = GH  + 98304;                // B*H
    float* P   = U   + 32768;                // B*H
    float* C0  = P   + 32768;                // B (pad 256)
    float* SC  = C0  + 256;                  // B*S
    float* CTX = SC  + 32768;                // B*H
    float* WON = CTX + 32768;                // B*OPN*S
    float* OP1 = WON + 131072;               // B*OPN*2H
    float* OP2 = OP1 + 262144;               // B*OPN*H
    float* OP3 = OP2 + 131072;               // B*OPN*H
    float* LG  = OP3 + 131072;               // B*O

    // bf16 pool at +8MB
    __bf16* bb16 = (__bf16*)((char*)d_ws + (8u << 20));
    __bf16* EBF   = bb16;                          // S*B*H   = 16777216
    __bf16* XW1T  = bb16 + 16777216;               // B*H*S   = 16777216
    __bf16* W1T   = bb16 + 33554432;               // H*H     = 262144
    __bf16* W2T   = bb16 + 33816576;               // 2H*H    = 524288
    __bf16* ABF   = bb16 + 34340864;               // B*S*S   = 16777216
    __bf16* H1BF  = bb16 + 51118080;               // B*S*H   = 16777216
    __bf16* Y3T   = bb16;                          // B*2H*S  = 33554432 (overlays EBF+XW1T, both dead)
    __bf16* W2WBF = bb16 + 67895296;               // B*S*2H  = 33554432

    const int T = 256;
    // ---- GRU ----
    k_gih <<<(Bc * H3 + T - 1) / T, T, 0, stream>>>(step, emb, lh, Wih, bih, Whh, bhh, GI, GH);
    k_hnew<<<(Bc * Hc + T - 1) / T, T, 0, stream>>>(GI, GH, lh, out_h);
    // ---- attention (algebraic rewrite: scores = (h@W)·enc + h·b) ----
    k_uP    <<<(Bc * Hc) / T, T, 0, stream>>>(out_h, attnW, gc1W, U, P);
    k_c0    <<<Bc, T, 0, stream>>>(out_h, attnb, C0);
    k_scores<<<(Bc * Sc) / T, T, 0, stream>>>(U, enc, C0, SC);
    k_softmax<<<Bc, T, 0, stream>>>(SC, out_aw);
    k_context<<<(Bc * Hc) / T, T, 0, stream>>>(out_aw, enc, CTX);
    k_concat <<<(Bc * Hc) / T, T, 0, stream>>>(out_h, CTX, cW, cb, out_concat);
    // ---- bf16 operand prep ----
    k_convE<<<16384, T, 0, stream>>>(enc, EBF, 4194304);
    k_w1t  <<<(Hc * Hc) / T, T, 0, stream>>>(gc1W, W1T);
    k_w2t  <<<(H2 * Hc) / T, T, 0, stream>>>(gc2W, W2T);
    k_adj  <<<16384, T, 0, stream>>>(ww, wex, ABF, 4194304);
    // ---- GCN via WMMA (block tile 128x128, wave tile 64x32) ----
    // G1: XW1t[b][n][s] = enc@W1b + aw*P   (M=S*B, N=H, K=H)
    k_gemm<0><<<dim3(Hc / 128, (Sc * Bc) / 128, 1), T, 0, stream>>>(
        EBF, W1T, Hc, Hc, Hc, 0, 0, XW1T, nullptr, out_aw, P);
    // G2: h1 = relu(adj @ XW1 + gc1_b)     (per b: M=S, N=H, K=S)
    k_gemm<1><<<dim3(Hc / 128, Sc / 128, Bc), T, 0, stream>>>(
        ABF, XW1T, Sc, Sc, Sc, (size_t)Sc * Sc, (size_t)Hc * Sc, H1BF, gc1b, nullptr, nullptr);
    // G3: Y3t[b][n][s] = (h1 @ gc2_W)^T    (M=B*S, N=2H, K=H)
    k_gemm<2><<<dim3(H2 / 128, (Bc * Sc) / 128, 1), T, 0, stream>>>(
        H1BF, W2T, Hc, Hc, Hc, 0, 0, Y3T, nullptr, nullptr, nullptr);
    // G4: w2w_pre = adj @ Y3 + gc2_b       (per b: M=S, N=2H, K=S)
    k_gemm<3><<<dim3(H2 / 128, Sc / 128, Bc), T, 0, stream>>>(
        ABF, Y3T, Sc, Sc, Sc, (size_t)Sc * Sc, (size_t)H2 * Sc, W2WBF, gc2b, nullptr, nullptr);
    // w2w = LN(w2w_pre) + w_all
    k_ln_w2w<<<Bc * Sc, T, 0, stream>>>(W2WBF, ng, nb, out_aw, out_h, enc);
    // ---- op heads ----
    k_won     <<<Bc * OPN, T, 0, stream>>>(wop, smask, WON);
    k_optrans1<<<(Bc * OPN * H2) / T, T, 0, stream>>>(WON, W2WBF, OP1);
    k_optrans2<<<(Bc * OPN * Hc) / T, T, 0, stream>>>(OP1, woW, wob, OP2);
    k_ln_small<<<Bc * OPN, T, 0, stream>>>(OP2, n1g, n1b, nullptr);
    k_oph     <<<(Bc * OPN * Hc) / T, T, 0, stream>>>(OP2, ops, otW, otb, OP3);
    k_ln_small<<<Bc * OPN, T, 0, stream>>>(OP3, n2g, n2b, ops);
    k_opout   <<<Bc * OPN, T, 0, stream>>>(out_concat, OP3, opsb, LG);
    k_nop     <<<(Bc * (Oc - OPN) + T - 1) / T, T, 0, stream>>>(out_concat, nopW, nopb, LG);
    k_logsm   <<<Bc, 128, 0, stream>>>(LG, out_logits);
}
